// CrossAttention_56968446214804
// MI455X (gfx1250) — compile-verified, matrix operand order checked by reading, not run on hardware
//
#include <hip/hip_runtime.h>
#include <hip/hip_bf16.h>
#include <math.h>

// ---------------------------------------------------------------------------
// Problem constants (from reference)
// ---------------------------------------------------------------------------
#define EMBED   128
#define HEADS   8
#define POINTS  4
#define H_SP    51
#define W_SP    102
#define NPIX    (H_SP * W_SP)      // 5202
#define NPIX_P  5216               // padded to 326*16
#define NQ      10000
#define NLVL    50
#define HD      16                 // EMBED / HEADS

typedef __attribute__((ext_vector_type(16))) _Float16 v16h;
typedef __attribute__((ext_vector_type(8)))  _Float16 v8h;
typedef __attribute__((ext_vector_type(8)))  float    v8f;

// ---------------------------------------------------------------------------
// WMMA fragment helpers (wave32, 16x16x32 f16 -> f32)
// A 16x32 (MxK): lanes 0-15 -> row M=lane,    K = {0..7, 16..23}
//                lanes 16-31 -> row M=lane-16, K = {8..15, 24..31}
// B 32x16 (KxN): mirrored (col = lane&15, same K striping)
// C/D 16x16 f32: VGPR v -> M = v + 8*(lane>=16), N = lane&15
// ---------------------------------------------------------------------------

// A fragment from f16 row-major tile (LDS): two aligned 16B vector loads.
__device__ inline v16h frag_A_f16(const _Float16* a, int lda) {
  int lane = threadIdx.x & 31;
  int row  = lane & 15;
  int kh   = (lane >> 4) << 3;
  v8h lo = *(const v8h*)(a + row * lda + kh);
  v8h hi = *(const v8h*)(a + row * lda + kh + 16);
  return __builtin_shufflevector(lo, hi, 0, 1, 2, 3, 4, 5, 6, 7,
                                 8, 9, 10, 11, 12, 13, 14, 15);
}

// A fragment from f32 row-major tile (global, rows known in-bounds):
// four aligned float4 loads + pack-convert. No guards, no branches.
__device__ inline v16h frag_A_f32(const float* a, int lda) {
  int lane = threadIdx.x & 31;
  int row  = lane & 15;
  int kh   = (lane >> 4) << 3;
  const float4* p0 = (const float4*)(a + row * lda + kh);
  const float4* p1 = (const float4*)(a + row * lda + kh + 16);
  float4 a0 = p0[0], a1 = p0[1];
  float4 b0 = p1[0], b1 = p1[1];
  v16h f;
  f[0]  = (_Float16)a0.x; f[1]  = (_Float16)a0.y;
  f[2]  = (_Float16)a0.z; f[3]  = (_Float16)a0.w;
  f[4]  = (_Float16)a1.x; f[5]  = (_Float16)a1.y;
  f[6]  = (_Float16)a1.z; f[7]  = (_Float16)a1.w;
  f[8]  = (_Float16)b0.x; f[9]  = (_Float16)b0.y;
  f[10] = (_Float16)b0.z; f[11] = (_Float16)b0.w;
  f[12] = (_Float16)b1.x; f[13] = (_Float16)b1.y;
  f[14] = (_Float16)b1.z; f[15] = (_Float16)b1.w;
  return f;
}

// B fragment from f32 column-strided weights (prep kernels only).
__device__ inline v16h frag_B_f32(const float* b, int ldb) {
  int lane = threadIdx.x & 31;
  int col  = lane & 15;
  int kh   = (lane >> 4) << 3;
  v16h f;
#pragma unroll
  for (int i = 0; i < 8; ++i) f[i]     = (_Float16)b[(kh + i) * ldb + col];
#pragma unroll
  for (int i = 0; i < 8; ++i) f[i + 8] = (_Float16)b[(kh + 16 + i) * ldb + col];
  return f;
}

#define WMMA_F16(A, B, C) \
  __builtin_amdgcn_wmma_f32_16x16x32_f16(false, (A), false, (B), (short)0, (C), false, false)

// ---------------------------------------------------------------------------
// Kernel 0a: zero-padded copy of value -> valP (5216 x 128)
// ---------------------------------------------------------------------------
__global__ void k_val_pad(const float* __restrict__ value, float* __restrict__ valP) {
  int t = blockIdx.x * blockDim.x + threadIdx.x;
  if (t < NPIX_P * 128) valP[t] = (t < NPIX * 128) ? value[t] : 0.0f;
}

// ---------------------------------------------------------------------------
// Kernel 0b: pre-pack W_out (f32 128x128) into WMMA B-fragment layout f16.
// packB[((ks*8 + nt)*32 + lane)*16 + i]   ks: k-slab (0..3), nt: n-tile (0..7)
// One thread per (ks, nt, lane) writes its 16 contiguous halves.
// ---------------------------------------------------------------------------
__global__ void k_wout_pack(const float* __restrict__ W_out, _Float16* __restrict__ packB) {
  int t = blockIdx.x * blockDim.x + threadIdx.x;   // 1024 total
  if (t >= 4 * 8 * 32) return;
  int lane = t & 31;
  int nt   = (t >> 5) & 7;
  int ks   = t >> 8;
  int col  = lane & 15;
  int kh   = (lane >> 4) << 3;
  int kb   = ks * 32;
  _Float16* dst = packB + (size_t)t * 16;
#pragma unroll
  for (int i = 0; i < 8; ++i)
    dst[i]     = (_Float16)W_out[(kb + kh + i) * 128 + nt * 16 + col];
#pragma unroll
  for (int i = 0; i < 8; ++i)
    dst[8 + i] = (_Float16)W_out[(kb + kh + 16 + i) * 128 + nt * 16 + col];
}

// ---------------------------------------------------------------------------
// Kernel 1: vproj = valP @ W_val + b_val   (5216 x 128, padded) via WMMA
// grid (326, 8) x 32 threads; no guards anywhere.
// ---------------------------------------------------------------------------
__global__ __launch_bounds__(32)
void k_val_gemm(const float* __restrict__ valP, const float* __restrict__ W_val,
                const float* __restrict__ b_val, float* __restrict__ vproj) {
  int m0 = blockIdx.x * 16;
  int n0 = blockIdx.y * 16;
  v8f acc = {};
#pragma unroll
  for (int kk = 0; kk < 128; kk += 32) {
    v16h a = frag_A_f32(valP + (size_t)m0 * 128 + kk, 128);
    v16h b = frag_B_f32(W_val + kk * 128 + n0, 128);
    acc = WMMA_F16(a, b, acc);
  }
  int lane = threadIdx.x & 31;
  int col  = lane & 15;
  int rb   = (lane >> 4) << 3;
  float bias = b_val[n0 + col];
#pragma unroll
  for (int v = 0; v < 8; ++v)
    vproj[(size_t)(m0 + rb + v) * 128 + n0 + col] = acc[v] + bias;
}

// ---------------------------------------------------------------------------
// Kernel 2: logits = query @ [W_off | W_attn] + bias   (10000 x 96) via WMMA
// grid (625, 6) x 32 threads; tiles 0..3 -> offsets, 4..5 -> attn logits
// ---------------------------------------------------------------------------
__global__ __launch_bounds__(32)
void k_qproj(const float* __restrict__ query,
             const float* __restrict__ W_off, const float* __restrict__ b_off,
             const float* __restrict__ W_attn, const float* __restrict__ b_attn,
             float* __restrict__ off_out, float* __restrict__ attnlog) {
  int m0 = blockIdx.x * 16;
  int n0 = blockIdx.y * 16;
  const float* B;
  const float* bias;
  int ldb;
  if (n0 < 64) { B = W_off  + n0;        ldb = 64; bias = b_off  + n0; }
  else         { B = W_attn + (n0 - 64); ldb = 32; bias = b_attn + (n0 - 64); }
  v8f acc = {};
#pragma unroll
  for (int kk = 0; kk < 128; kk += 32) {
    v16h a = frag_A_f32(query + (size_t)m0 * 128 + kk, 128);
    v16h b = frag_B_f32(B + kk * ldb, ldb);
    acc = WMMA_F16(a, b, acc);
  }
  int lane = threadIdx.x & 31;
  int col  = lane & 15;
  int rb   = (lane >> 4) << 3;
  float bv = bias[col];
#pragma unroll
  for (int v = 0; v < 8; ++v) {
    int row = rb + v;
    float val = acc[v] + bv;
    if (n0 < 64) off_out[(size_t)(m0 + row) * 64 + n0 + col] = val;
    else         attnlog[(size_t)(m0 + row) * 32 + (n0 - 64) + col] = val;
  }
}

// ---------------------------------------------------------------------------
// Kernel 3: softmax over POINTS=4 per (query, head)
// ---------------------------------------------------------------------------
__global__ void k_attn_softmax(const float* __restrict__ logits, float* __restrict__ aw) {
  int t = blockIdx.x * blockDim.x + threadIdx.x;   // (q*8 + h)
  if (t >= NQ * HEADS) return;
  const float4 l = ((const float4*)logits)[t];     // == q*32 + h*4
  float m = fmaxf(fmaxf(l.x, l.y), fmaxf(l.z, l.w));
  float ea = __expf(l.x - m), eb = __expf(l.y - m);
  float ec = __expf(l.z - m), ed = __expf(l.w - m);
  float inv = 1.0f / (ea + eb + ec + ed);
  float4 o; o.x = ea * inv; o.y = eb * inv; o.z = ec * inv; o.w = ed * inv;
  ((float4*)aw)[t] = o;
}

// ---------------------------------------------------------------------------
// bilinear corner gather: 16 contiguous floats (one head slice), 4x float4
// ---------------------------------------------------------------------------
__device__ inline void gather16(const float* __restrict__ p, float w, float* acc) {
  const float4* p4 = (const float4*)p;
#pragma unroll
  for (int i = 0; i < 4; ++i) {
    float4 v = p4[i];
    acc[i * 4 + 0] += w * v.x;
    acc[i * 4 + 1] += w * v.y;
    acc[i * 4 + 2] += w * v.z;
    acc[i * 4 + 3] += w * v.w;
  }
}

// ---------------------------------------------------------------------------
// Kernel 5: main fused loop. 625 blocks x 128 threads, 16 queries/block.
// Streams 50 levels with online softmax + argmax over levels.
// ---------------------------------------------------------------------------
__global__ __launch_bounds__(128)
void k_main(const float* __restrict__ query, const float* __restrict__ ref,
            const float* __restrict__ vproj, const float* __restrict__ off,
            const float* __restrict__ aw, const _Float16* __restrict__ packB,
            const float* __restrict__ b_out, const float* __restrict__ W_proj,
            const float* __restrict__ b_proj,
            float* __restrict__ out, int* __restrict__ argout) {
  __shared__ float    q_t[16][128];      // query tile (residuals)
  __shared__ float    off_t[16][64];     // offsets
  __shared__ float    aw_t[16][32];      // attention weights
  __shared__ _Float16 msda_h[16 * 128];  // per-level MSDA output (A operand)
  __shared__ float    g_t[16][128];      // per-level g = msda@W_out+b+q
  __shared__ float    Nacc[16][128];     // online softmax numerator
  __shared__ float    wproj_s[128];
  __shared__ float    Zs[16], Ms[16], Ss[16];
  __shared__ int      Ai[16];

  int tid = threadIdx.x;
  int q0  = blockIdx.x * 16;

  for (int i = tid; i < 16 * 128; i += 128) {
    q_t[i >> 7][i & 127]  = query[(size_t)q0 * 128 + i];
    Nacc[i >> 7][i & 127] = 0.0f;
  }
  for (int i = tid; i < 16 * 64; i += 128) off_t[i >> 6][i & 63] = off[(size_t)q0 * 64 + i];
  for (int i = tid; i < 16 * 32; i += 128) aw_t[i >> 5][i & 31]  = aw[(size_t)q0 * 32 + i];
  wproj_s[tid] = W_proj[tid];
  if (tid < 16) { Zs[tid] = 0.0f; Ms[tid] = -1e30f; Ai[tid] = 0; }
  float bproj = b_proj[0];
  __syncthreads();

  int lane  = tid & 31;
  int wave  = tid >> 5;
  int col16 = lane & 15;
  int rb    = (lane >> 4) << 3;
  int nb    = wave * 32;
  int sq    = tid >> 3;                  // owned query for reductions
  int sch   = (tid & 7) << 4;            // owned 16-channel slice

  const v16h* pb = (const v16h*)packB;   // fragment-layout W_out

  for (int l = 0; l < NLVL; ++l) {
    if (tid < 16) Ss[tid] = bproj;

    // ---- deformable sampling: one thread per (query, head) ----
    {
      int q = tid >> 3, h = tid & 7;
      const float* rp = ref + (size_t)(l * NQ + q0 + q) * 2;
      float rx = rp[0], ry = rp[1];
      const float* vh = vproj + h * HD;
      float acc[HD];
#pragma unroll
      for (int c = 0; c < HD; ++c) acc[c] = 0.0f;
#pragma unroll
      for (int p = 0; p < POINTS; ++p) {
        float w  = aw_t[q][h * 4 + p];
        float lx = rx + off_t[q][(h * 4 + p) * 2 + 0] * (1.0f / (float)W_SP);
        float ly = ry + off_t[q][(h * 4 + p) * 2 + 1] * (1.0f / (float)H_SP);
        float px = lx * (float)W_SP - 0.5f;
        float py = ly * (float)H_SP - 0.5f;
        float fx = floorf(px), fy = floorf(py);
        int   x0 = (int)fx, y0 = (int)fy;
        float dx = px - fx, dy = py - fy;
        float w00 = w * (1.0f - dx) * (1.0f - dy);
        float w01 = w * dx * (1.0f - dy);
        float w10 = w * (1.0f - dx) * dy;
        float w11 = w * dx * dy;
        bool vx0 = (x0 >= 0) & (x0 < W_SP);
        bool vx1 = (x0 + 1 >= 0) & (x0 + 1 < W_SP);
        bool vy0 = (y0 >= 0) & (y0 < H_SP);
        bool vy1 = (y0 + 1 >= 0) & (y0 + 1 < H_SP);
        if (vx0 & vy0) gather16(vh + (size_t)(y0 * W_SP + x0) * 128, w00, acc);
        if (vx1 & vy0) gather16(vh + (size_t)(y0 * W_SP + x0 + 1) * 128, w01, acc);
        if (vx0 & vy1) gather16(vh + (size_t)((y0 + 1) * W_SP + x0) * 128, w10, acc);
        if (vx1 & vy1) gather16(vh + (size_t)((y0 + 1) * W_SP + x0 + 1) * 128, w11, acc);
      }
#pragma unroll
      for (int c = 0; c < HD; ++c)
        msda_h[q * 128 + h * HD + c] = (_Float16)acc[c];
    }
    __syncthreads();

    // ---- g = msda @ W_out (+ b_out + query); 4 waves x 32 columns ----
    {
      v8f acc0 = {}, acc1 = {};
      int nt0 = wave * 2;                // n-tiles owned by this wave
#pragma unroll
      for (int ks = 0; ks < 4; ++ks) {
        v16h a  = frag_A_f16(msda_h + ks * 32, 128);
        v16h b0 = pb[((ks * 8) + nt0) * 32 + lane];        // single 32B load
        v16h b1 = pb[((ks * 8) + nt0 + 1) * 32 + lane];
        acc0 = WMMA_F16(a, b0, acc0);
        acc1 = WMMA_F16(a, b1, acc1);
      }
      int n0c = nb + col16, n1c = nb + 16 + col16;
      float bo0 = b_out[n0c], bo1 = b_out[n1c];
#pragma unroll
      for (int v = 0; v < 8; ++v) {
        int row = rb + v;
        g_t[row][n0c] = acc0[v] + bo0 + q_t[row][n0c];
        g_t[row][n1c] = acc1[v] + bo1 + q_t[row][n1c];
      }
    }
    __syncthreads();

    // ---- level score: s = g . W_proj + b_proj ----
    {
      float part = 0.0f;
#pragma unroll
      for (int c = 0; c < 16; ++c) part += g_t[sq][sch + c] * wproj_s[sch + c];
      atomicAdd(&Ss[sq], part);
    }
    __syncthreads();

    // ---- online softmax over levels + argmax ----
    {
      float s    = Ss[sq];
      float mold = Ms[sq];
      float mnew = fmaxf(s, mold);
      float fo   = __expf(mold - mnew);
      float fn   = __expf(s - mnew);
#pragma unroll
      for (int c = 0; c < 16; ++c)
        Nacc[sq][sch + c] = Nacc[sq][sch + c] * fo + g_t[sq][sch + c] * fn;
      if ((tid & 7) == 0) {
        Zs[sq] = Zs[sq] * fo + fn;
        if (s > mold) { Ms[sq] = mnew; Ai[sq] = l; }
      }
    }
    __syncthreads();
  }

  // ---- final: out = N/Z + query ; argmax index ----
  {
    float invZ = 1.0f / Zs[sq];
#pragma unroll
    for (int c = 0; c < 16; ++c)
      out[(size_t)(q0 + sq) * 128 + sch + c] = Nacc[sq][sch + c] * invZ + q_t[sq][sch + c];
    if ((tid & 7) == 0) argout[q0 + sq] = Ai[sq];
  }
}

// ---------------------------------------------------------------------------
// Launch
// ---------------------------------------------------------------------------
extern "C" void kernel_launch(void* const* d_in, const int* in_sizes, int n_in,
                              void* d_out, int out_size, void* d_ws, size_t ws_size,
                              hipStream_t stream) {
  const float* query  = (const float*)d_in[0];
  const float* value  = (const float*)d_in[1];
  const float* refpts = (const float*)d_in[2];
  const float* W_off  = (const float*)d_in[3];
  const float* b_off  = (const float*)d_in[4];
  const float* W_attn = (const float*)d_in[5];
  const float* b_attn = (const float*)d_in[6];
  const float* W_val  = (const float*)d_in[7];
  const float* b_val  = (const float*)d_in[8];
  const float* W_out  = (const float*)d_in[9];
  const float* b_out  = (const float*)d_in[10];
  const float* W_proj = (const float*)d_in[11];
  const float* b_proj = (const float*)d_in[12];

  float* out    = (float*)d_out;
  int*   argout = (int*)((float*)d_out + (size_t)NQ * 128);

  // workspace layout (256B-aligned slices)
  size_t cur = 0;
  auto take = [&](size_t bytes) {
    void* p = (char*)d_ws + cur;
    cur = (cur + bytes + 255) & ~(size_t)255;
    return p;
  };
  float*    valP    = (float*)take((size_t)NPIX_P * 128 * 4);
  float*    vproj   = (float*)take((size_t)NPIX_P * 128 * 4);
  float*    offb    = (float*)take((size_t)NQ * 64 * 4);
  float*    attnlog = (float*)take((size_t)NQ * 32 * 4);
  float*    awb     = (float*)take((size_t)NQ * 32 * 4);
  _Float16* packB   = (_Float16*)take((size_t)128 * 128 * 2);

  k_val_pad<<<(NPIX_P * 128 + 255) / 256, 256, 0, stream>>>(value, valP);
  k_wout_pack<<<4, 256, 0, stream>>>(W_out, packB);
  k_val_gemm<<<dim3(NPIX_P / 16, 8), 32, 0, stream>>>(valP, W_val, b_val, vproj);
  k_qproj<<<dim3(NQ / 16, 6), 32, 0, stream>>>(query, W_off, b_off, W_attn, b_attn,
                                               offb, attnlog);
  k_attn_softmax<<<(NQ * HEADS + 255) / 256, 256, 0, stream>>>(attnlog, awb);
  k_main<<<NQ / 16, 128, 0, stream>>>(query, refpts, vproj, offb, awb, packB,
                                      b_out, W_proj, b_proj, out, argout);
}
